// MultiHeadAttention_14499809592049
// MI455X (gfx1250) — matrix operand order repaired and checked by reading, not compiled, not run on hardware
//
#include <hip/hip_runtime.h>
#include <hip/hip_bf16.h>

#define EMB 1024
#define SEQ 2048
#define NB 4
#define NH 16
#define DH 64
#define MROWS (NB * SEQ)  // 8192

typedef __attribute__((ext_vector_type(16))) __bf16 v16bf;
typedef __attribute__((ext_vector_type(8)))  float  v8f;
typedef __attribute__((ext_vector_type(4)))  unsigned v4u;
typedef __attribute__((ext_vector_type(4)))  int      v4i;
typedef __attribute__((ext_vector_type(8)))  int      v8i;

union FragBF { uint4 u[2]; v16bf v; };

__device__ __forceinline__ unsigned short f2bf(float f) {
  return (unsigned short)((__builtin_bit_cast(unsigned, f) + 0x8000u) >> 16);
}
// pack two fp32 -> packed bf16 pair with one v_perm_b32
__device__ __forceinline__ unsigned pk2(float a, float b) {
  unsigned ra = __builtin_bit_cast(unsigned, a) + 0x8000u;
  unsigned rb_ = __builtin_bit_cast(unsigned, b) + 0x8000u;
  return __builtin_amdgcn_perm(rb_, ra, 0x07060302u);
}
__device__ __forceinline__ v8f vzero() {
  v8f z;
#pragma unroll
  for (int i = 0; i < 8; ++i) z[i] = 0.0f;
  return z;
}

// ---------------------------------------------------------------------------
// Tensor Data Mover: 2D bf16 tile (tile_w x tile_h elements) global -> LDS.
// D# packing per CDNA5 ISA. Issue from ONE wave.
// ---------------------------------------------------------------------------
__device__ __forceinline__ void tdm_load_2d_bf16(const void* gptr, const void* ldsptr,
                                                 unsigned tile_w, unsigned tile_h,
                                                 unsigned row_stride_elems,
                                                 unsigned tensor_w, unsigned tensor_h) {
  unsigned long long ga = (unsigned long long)(size_t)gptr;
  unsigned lds_off = (unsigned)(unsigned long long)(size_t)ldsptr;
  v4u g0;
  g0[0] = 1u;                                                 // count=1
  g0[1] = lds_off;                                            // lds_addr
  g0[2] = (unsigned)(ga & 0xFFFFFFFFu);                       // global_addr lo
  g0[3] = (unsigned)((ga >> 32) & 0x01FFFFFFu) | (2u << 30);  // addr hi | type=2
  v8i g1;
  g1[0] = (int)(1u << 16);                                    // data_size = 2B
  g1[1] = (int)((tensor_w & 0xFFFFu) << 16);
  g1[2] = (int)(((tensor_w >> 16) & 0xFFFFu) | ((tensor_h & 0xFFFFu) << 16));
  g1[3] = (int)(((tensor_h >> 16) & 0xFFFFu) | ((tile_w & 0xFFFFu) << 16));
  g1[4] = (int)(tile_h & 0xFFFFu);
  g1[5] = (int)row_stride_elems;
  g1[6] = 0;
  g1[7] = 0;
  v4i z4 = {0, 0, 0, 0};
#if defined(__clang_major__) && (__clang_major__ >= 23)
  v8i z8 = {0, 0, 0, 0, 0, 0, 0, 0};
  __builtin_amdgcn_tensor_load_to_lds(g0, g1, z4, z4, z8, 0);
#else
  __builtin_amdgcn_tensor_load_to_lds(g0, g1, z4, z4, 0);
#endif
}

// ---------------------------------------------------------------------------
// Y[m,n] = sum_k X[m,k]*W[n,k] + bias[n] -> bf16 QKV layout. Block 128x128,
// 8 waves x (32x64), double-buffered LDS, ONE barrier per 32-deep k-step,
// coalesced b128 epilogue via LDS restage.
// ---------------------------------------------------------------------------
__global__ __launch_bounds__(256) void proj_qkv_kernel(
    const float* __restrict__ X, const float* __restrict__ W,
    const float* __restrict__ bias, unsigned short* __restrict__ out) {
  __shared__ alignas(16) unsigned short Sm[16384];  // 32KB: As[2](8KB) Bs[2](8KB)

  const int tid = threadIdx.x;
  const int lane = tid & 31;
  const int wv = tid >> 5;
  const int mt = wv >> 1;
  const int nh = wv & 1;
  const int hi = lane >> 4;
  const int ln = lane & 15;
  const int m0 = blockIdx.x * 128;
  const int n0 = blockIdx.y * 128;
  const int r8 = tid >> 3, c8 = tid & 7;

  v8f acc[2][4];
#pragma unroll
  for (int t = 0; t < 4; ++t) {
    float bvv = bias[n0 + nh * 64 + t * 16 + ln];
    v8f a;
#pragma unroll
    for (int i = 0; i < 8; ++i) a[i] = bvv;
    acc[0][t] = a;
    acc[1][t] = a;
  }

  float4 ra[4], rb[4];
#pragma unroll
  for (int i = 0; i < 4; ++i) {
    int r = r8 + i * 32;
    ra[i] = *(const float4*)(X + (size_t)(m0 + r) * EMB + c8 * 4);
    rb[i] = *(const float4*)(W + (size_t)(n0 + r) * EMB + c8 * 4);
  }

  for (int it = 0; it < EMB / 32; ++it) {
    const int cur = it & 1;
    unsigned* Au = (unsigned*)(Sm + cur * 4096);
    unsigned* Bu = (unsigned*)(Sm + 8192 + cur * 4096);
#pragma unroll
    for (int i = 0; i < 4; ++i) {
      int r = r8 + i * 32;
      Au[r * 16 + c8 * 2]     = pk2(ra[i].x, ra[i].y);
      Au[r * 16 + c8 * 2 + 1] = pk2(ra[i].z, ra[i].w);
      Bu[r * 16 + c8 * 2]     = pk2(rb[i].x, rb[i].y);
      Bu[r * 16 + c8 * 2 + 1] = pk2(rb[i].z, rb[i].w);
    }
    if (it + 1 < EMB / 32) {
      int k1 = (it + 1) * 32;
#pragma unroll
      for (int i = 0; i < 4; ++i) {
        int r = r8 + i * 32;
        ra[i] = *(const float4*)(X + (size_t)(m0 + r) * EMB + k1 + c8 * 4);
        rb[i] = *(const float4*)(W + (size_t)(n0 + r) * EMB + k1 + c8 * 4);
      }
    }
    __syncthreads();

    const uint4* As4 = (const uint4*)(Sm + cur * 4096);
    const uint4* Bs4 = (const uint4*)(Sm + 8192 + cur * 4096);
    FragBF af[2];
#pragma unroll
    for (int s = 0; s < 2; ++s) {
      int arow = (mt * 32 + s * 16 + ln) * 4;
      af[s].u[0] = As4[arow + hi];
      af[s].u[1] = As4[arow + 2 + hi];
    }
    FragBF bf[4];
#pragma unroll
    for (int t = 0; t < 4; ++t) {
      int brow = (nh * 64 + t * 16 + ln) * 4;
      bf[t].u[0] = Bs4[brow + hi];
      bf[t].u[1] = Bs4[brow + 2 + hi];
    }
#pragma unroll
    for (int t = 0; t < 4; ++t)
#pragma unroll
      for (int s = 0; s < 2; ++s)
        acc[s][t] = __builtin_amdgcn_wmma_f32_16x16x32_bf16(
            false, af[s].v, false, bf[t].v, (short)0, acc[s][t], false, false);
    // no trailing barrier: next iteration writes the other buffer
  }

  // epilogue: restage 128x128 bf16 tile in LDS, then coalesced b128 stores
  __syncthreads();
#pragma unroll
  for (int s = 0; s < 2; ++s)
#pragma unroll
    for (int t = 0; t < 4; ++t)
#pragma unroll
      for (int r = 0; r < 8; ++r) {
        int ml = mt * 32 + s * 16 + r + hi * 8;
        int el = nh * 64 + t * 16 + ln;
        Sm[ml * 128 + el] = f2bf(acc[s][t][r]);
      }
  __syncthreads();
#pragma unroll
  for (int j = 0; j < 8; ++j) {
    int idx = tid + j * 256;            // 2048 uint4
    int row = idx >> 4, c4 = idx & 15;  // tile row, 8-elem column chunk
    int m = m0 + row;
    int b = m >> 11, sq = m & (SEQ - 1);
    int e = n0 + c4 * 8;
    int h = e >> 6, d = e & 63;
    *(uint4*)(out + (((size_t)(b * NH + h)) * SEQ + sq) * DH + d) =
        ((const uint4*)Sm)[idx];
  }
}

// ---------------------------------------------------------------------------
// Flash attention: grid (SEQ/128, NB*NH). 64-key KV tiles, K via TDM
// (double-buffered), V register-transposed (double-buffered), online softmax,
// P via per-wave LDS, coalesced epilogue through restaged LDS.
// ---------------------------------------------------------------------------
__global__ __launch_bounds__(256) void attn_kernel(
    const unsigned short* __restrict__ Qw, const unsigned short* __restrict__ Kw,
    const unsigned short* __restrict__ Vw, unsigned short* __restrict__ Ow) {
  __shared__ alignas(16) unsigned short Ks[2][64 * 64];   // [key][dh]
  __shared__ alignas(16) unsigned short Vt[2][64 * 64];   // [dh][key]
  __shared__ alignas(16) unsigned short Ps[8 * 16 * 64];  // per-wave P

  const int tid = threadIdx.x;
  const int lane = tid & 31;
  const int wv = tid >> 5;
  const int hi = lane >> 4;
  const int ln = lane & 15;
  const int qt = blockIdx.x;
  const int bh = blockIdx.y;
  const size_t base = (size_t)bh * SEQ * DH;

  const uint4* Qrow = (const uint4*)(Qw + base + (size_t)(qt * 128 + wv * 16 + ln) * DH);
  FragBF qa0, qa1;
  qa0.u[0] = Qrow[hi];     qa0.u[1] = Qrow[2 + hi];
  qa1.u[0] = Qrow[4 + hi]; qa1.u[1] = Qrow[6 + hi];

  float mrow[8], lrow[8];
#pragma unroll
  for (int r = 0; r < 8; ++r) { mrow[r] = -3.0e38f; lrow[r] = 0.0f; }
  v8f o[4];
#pragma unroll
  for (int nt = 0; nt < 4; ++nt) o[nt] = vzero();

  const float sscale = 0.125f * 1.44269504088896340736f;  // 1/sqrt(Dh)*log2(e)
  unsigned short* pw = Ps + wv * 1024;
  const uint4* P4 = (const uint4*)pw;
  const int r8 = tid >> 3, c8 = tid & 7;

  uint4 rv[2];  // V staging regs (64 keys x 64 dh = 512 uint4, 2/thread)
#pragma unroll
  for (int j = 0; j < 2; ++j) {
    int idx = tid + j * 256;
    rv[j] = *(const uint4*)(Vw + base + (size_t)(idx >> 3) * DH + (idx & 7) * 8);
  }
  if (tid < 32)
    tdm_load_2d_bf16(Kw + base, Ks[0], DH, 64, DH, DH, 64);

  for (int it = 0; it < SEQ / 64; ++it) {
    const int cur = it & 1;
    unsigned short* vt = Vt[cur];
#pragma unroll
    for (int j = 0; j < 2; ++j) {  // transpose-store V tile (it)
      int idx = tid + j * 256;
      int r = idx >> 3, c = idx & 7;
      const unsigned short* pv = (const unsigned short*)&rv[j];
#pragma unroll
      for (int q = 0; q < 8; ++q) vt[(c * 8 + q) * 64 + r] = pv[q];
    }
    if (it + 1 < SEQ / 64) {  // fetch next V tile into regs
#pragma unroll
      for (int j = 0; j < 2; ++j) {
        int idx = tid + j * 256;
        rv[j] = *(const uint4*)(Vw + base + (size_t)((it + 1) * 64 + (idx >> 3)) * DH +
                                (idx & 7) * 8);
      }
    }
    if (tid < 32) __builtin_amdgcn_s_wait_tensorcnt(0);
    __syncthreads();
    if (it + 1 < SEQ / 64 && tid < 32)
      tdm_load_2d_bf16(Kw + base + (size_t)(it + 1) * 64 * DH, Ks[cur ^ 1],
                       DH, 64, DH, DH, 64);

    const uint4* Ks4 = (const uint4*)Ks[cur];
    const uint4* Vt4 = (const uint4*)vt;

    // S = Q @ K^T * scale : four 16-key tiles
    v8f sc[4];
#pragma unroll
    for (int t = 0; t < 4; ++t) {
      int krow = (t * 16 + ln) * 8;
      FragBF kf0, kf1;
      kf0.u[0] = Ks4[krow + hi];     kf0.u[1] = Ks4[krow + 2 + hi];
      kf1.u[0] = Ks4[krow + 4 + hi]; kf1.u[1] = Ks4[krow + 6 + hi];
      v8f z = vzero();
      z = __builtin_amdgcn_wmma_f32_16x16x32_bf16(false, qa0.v, false, kf0.v, (short)0, z, false, false);
      z = __builtin_amdgcn_wmma_f32_16x16x32_bf16(false, qa1.v, false, kf1.v, (short)0, z, false, false);
      sc[t] = z * sscale;
    }

    float alpha[8];
#pragma unroll
    for (int r = 0; r < 8; ++r) {
      float tmax = fmaxf(fmaxf(sc[0][r], sc[1][r]), fmaxf(sc[2][r], sc[3][r]));
#pragma unroll
      for (int msk = 1; msk < 16; msk <<= 1)
        tmax = fmaxf(tmax, __shfl_xor(tmax, msk, 32));
      float mn = fmaxf(mrow[r], tmax);
      float p0 = exp2f(sc[0][r] - mn), p1 = exp2f(sc[1][r] - mn);
      float p2 = exp2f(sc[2][r] - mn), p3 = exp2f(sc[3][r] - mn);
      float rs = (p0 + p1) + (p2 + p3);
#pragma unroll
      for (int msk = 1; msk < 16; msk <<= 1)
        rs += __shfl_xor(rs, msk, 32);
      float a = exp2f(mrow[r] - mn);
      lrow[r] = lrow[r] * a + rs;
      mrow[r] = mn;
      alpha[r] = a;
      sc[0][r] = p0; sc[1][r] = p1; sc[2][r] = p2; sc[3][r] = p3;
    }
#pragma unroll
    for (int nt = 0; nt < 4; ++nt)
#pragma unroll
      for (int r = 0; r < 8; ++r) o[nt][r] *= alpha[r];

    // P: C-layout -> per-wave LDS -> two A-fragments (keys 0..31 / 32..63)
#pragma unroll
    for (int r = 0; r < 8; ++r) {
      int m = r + hi * 8;
#pragma unroll
      for (int t = 0; t < 4; ++t) pw[m * 64 + t * 16 + ln] = f2bf(sc[t][r]);
    }
    asm volatile("s_wait_dscnt 0" ::: "memory");

    FragBF pa0, pa1;
    pa0.u[0] = P4[ln * 8 + hi];     pa0.u[1] = P4[ln * 8 + 2 + hi];
    pa1.u[0] = P4[ln * 8 + 4 + hi]; pa1.u[1] = P4[ln * 8 + 6 + hi];
#pragma unroll
    for (int nt = 0; nt < 4; ++nt) {
      int vrow = (nt * 16 + ln) * 8;
      FragBF vb0, vb1;
      vb0.u[0] = Vt4[vrow + hi];     vb0.u[1] = Vt4[vrow + 2 + hi];
      vb1.u[0] = Vt4[vrow + 4 + hi]; vb1.u[1] = Vt4[vrow + 6 + hi];
      o[nt] = __builtin_amdgcn_wmma_f32_16x16x32_bf16(false, pa0.v, false, vb0.v, (short)0, o[nt], false, false);
      o[nt] = __builtin_amdgcn_wmma_f32_16x16x32_bf16(false, pa1.v, false, vb1.v, (short)0, o[nt], false, false);
    }
    // no trailing barrier: buffers are double-buffered
  }

  // epilogue: restage 128x64 bf16 in LDS (reuse Ks), coalesced b128 stores
  __syncthreads();
  unsigned short* stage = &Ks[0][0];  // 16KB
#pragma unroll
  for (int r = 0; r < 8; ++r) {
    float inv = 1.0f / lrow[r];
    int m = wv * 16 + r + hi * 8;
#pragma unroll
    for (int nt = 0; nt < 4; ++nt)
      stage[m * 64 + nt * 16 + ln] = f2bf(o[nt][r] * inv);
  }
  __syncthreads();
  const int b = bh >> 4, h = bh & 15;
#pragma unroll
  for (int j = 0; j < 4; ++j) {
    int idx = tid + j * 256;  // 1024 uint4
    int sr = idx >> 3, c = idx & 7;
    *(uint4*)(Ow + ((size_t)b * SEQ + qt * 128 + sr) * EMB + h * 64 + c * 8) =
        ((const uint4*)stage)[idx];
  }
}

// ---------------------------------------------------------------------------
// out = O @ Wo^T + bo. A tile via TDM (double-buffered), B staged through
// regs, one barrier per k-step, fp32 output.
// ---------------------------------------------------------------------------
__global__ __launch_bounds__(256) void out_proj_kernel(
    const unsigned short* __restrict__ Ain, const float* __restrict__ W,
    const float* __restrict__ bias, float* __restrict__ out) {
  __shared__ alignas(16) unsigned short As[2][128 * 32];
  __shared__ alignas(16) unsigned short Bs[2][128 * 32];

  const int tid = threadIdx.x;
  const int lane = tid & 31;
  const int wv = tid >> 5;
  const int mt = wv >> 1;
  const int nh = wv & 1;
  const int hi = lane >> 4;
  const int ln = lane & 15;
  const int m0 = blockIdx.x * 128;
  const int n0 = blockIdx.y * 128;
  const int r8 = tid >> 3, c8 = tid & 7;

  v8f acc[2][4];
#pragma unroll
  for (int t = 0; t < 4; ++t) {
    float bvv = bias[n0 + nh * 64 + t * 16 + ln];
    v8f a;
#pragma unroll
    for (int i = 0; i < 8; ++i) a[i] = bvv;
    acc[0][t] = a;
    acc[1][t] = a;
  }

  float4 rb[4];
#pragma unroll
  for (int i = 0; i < 4; ++i)
    rb[i] = *(const float4*)(W + (size_t)(n0 + r8 + i * 32) * EMB + c8 * 4);
  if (tid < 32)
    tdm_load_2d_bf16(Ain + (size_t)m0 * EMB, As[0], 32, 128, EMB, EMB, 128);

  for (int it = 0; it < EMB / 32; ++it) {
    const int cur = it & 1;
    unsigned* Bu = (unsigned*)Bs[cur];
#pragma unroll
    for (int i = 0; i < 4; ++i) {
      int r = r8 + i * 32;
      Bu[r * 16 + c8 * 2]     = pk2(rb[i].x, rb[i].y);
      Bu[r * 16 + c8 * 2 + 1] = pk2(rb[i].z, rb[i].w);
    }
    if (it + 1 < EMB / 32) {
      int k1 = (it + 1) * 32;
#pragma unroll
      for (int i = 0; i < 4; ++i)
        rb[i] = *(const float4*)(W + (size_t)(n0 + r8 + i * 32) * EMB + k1 + c8 * 4);
    }
    if (tid < 32) __builtin_amdgcn_s_wait_tensorcnt(0);
    __syncthreads();
    if (it + 1 < EMB / 32 && tid < 32)
      tdm_load_2d_bf16(Ain + (size_t)m0 * EMB + (it + 1) * 32, As[cur ^ 1],
                       32, 128, EMB, EMB, 128);

    const uint4* As4 = (const uint4*)As[cur];
    const uint4* Bs4 = (const uint4*)Bs[cur];
    FragBF af[2];
#pragma unroll
    for (int s = 0; s < 2; ++s) {
      int arow = (mt * 32 + s * 16 + ln) * 4;
      af[s].u[0] = As4[arow + hi];
      af[s].u[1] = As4[arow + 2 + hi];
    }
    FragBF bf[4];
#pragma unroll
    for (int t = 0; t < 4; ++t) {
      int brow = (nh * 64 + t * 16 + ln) * 4;
      bf[t].u[0] = Bs4[brow + hi];
      bf[t].u[1] = Bs4[brow + 2 + hi];
    }
#pragma unroll
    for (int t = 0; t < 4; ++t)
#pragma unroll
      for (int s = 0; s < 2; ++s)
        acc[s][t] = __builtin_amdgcn_wmma_f32_16x16x32_bf16(
            false, af[s].v, false, bf[t].v, (short)0, acc[s][t], false, false);
  }

#pragma unroll
  for (int s = 0; s < 2; ++s)
#pragma unroll
    for (int t = 0; t < 4; ++t)
#pragma unroll
      for (int r = 0; r < 8; ++r) {
        int m = m0 + mt * 32 + s * 16 + r + hi * 8;
        int e = n0 + nh * 64 + t * 16 + ln;
        out[(size_t)m * EMB + e] = acc[s][t][r];
      }
}

extern "C" void kernel_launch(void* const* d_in, const int* in_sizes, int n_in,
                              void* d_out, int out_size, void* d_ws, size_t ws_size,
                              hipStream_t stream) {
  (void)in_sizes; (void)n_in; (void)out_size; (void)ws_size;
  const float* query = (const float*)d_in[0];
  const float* key_  = (const float*)d_in[1];
  const float* value = (const float*)d_in[2];
  const float* Wq = (const float*)d_in[3];
  const float* bq = (const float*)d_in[4];
  const float* Wk = (const float*)d_in[5];
  const float* bk = (const float*)d_in[6];
  const float* Wv = (const float*)d_in[7];
  const float* bv = (const float*)d_in[8];
  const float* Wo = (const float*)d_in[9];
  const float* bo = (const float*)d_in[10];

  const size_t elems = (size_t)MROWS * EMB;  // 8M bf16 each
  unsigned short* Qw = (unsigned short*)d_ws;
  unsigned short* Kw = Qw + elems;
  unsigned short* Vw = Kw + elems;
  unsigned short* Ow = Vw + elems;  // 64 MB workspace total

  dim3 gp(MROWS / 128, EMB / 128), bp(256);
  proj_qkv_kernel<<<gp, bp, 0, stream>>>(query, Wq, bq, Qw);
  proj_qkv_kernel<<<gp, bp, 0, stream>>>(key_,  Wk, bk, Kw);
  proj_qkv_kernel<<<gp, bp, 0, stream>>>(value, Wv, bv, Vw);
  attn_kernel<<<dim3(SEQ / 128, NB * NH), 256, 0, stream>>>(Qw, Kw, Vw, Ow);
  out_proj_kernel<<<gp, bp, 0, stream>>>(Ow, Wo, bo, (float*)d_out);
}
